// BigHeteroGNN_86535001080513
// MI455X (gfx1250) — matrix operand order repaired and checked by reading, not compiled, not run on hardware
//
#include <hip/hip_runtime.h>

// ---------------------------------------------------------------------------
// Types for CDNA5 WMMA (wave32)
// ---------------------------------------------------------------------------
typedef __bf16 bf16;
typedef __attribute__((ext_vector_type(16))) __bf16 v16bf;
typedef __attribute__((ext_vector_type(8)))  __bf16 v8bf;
typedef __attribute__((ext_vector_type(8)))  float  v8f;

#define HID 128
#define HEADS 4
#define DH 32

__device__ __forceinline__ bf16 f2bf(float f) {
  unsigned u = __float_as_uint(f);
  u = u + 0x7FFFu + ((u >> 16) & 1u);   // round-to-nearest-even
  unsigned short s = (unsigned short)(u >> 16);
  bf16 b;
  __builtin_memcpy(&b, &s, sizeof(s));
  return b;
}

__device__ __forceinline__ float decode_maxenc(unsigned int e) {
  unsigned b = (e & 0x80000000u) ? (e & 0x7FFFFFFFu) : ~e;
  return __uint_as_float(b);
}

// ---------------------------------------------------------------------------
// Elementwise helpers
// ---------------------------------------------------------------------------
__global__ void cvt_f32_bf16_kernel(const float* __restrict__ in, bf16* __restrict__ out, long n) {
  long i = (long)blockIdx.x * blockDim.x + threadIdx.x;
  if (i < n) out[i] = f2bf(in[i]);
}

__global__ void fill_u32_kernel(unsigned int* __restrict__ p, unsigned int v, long n) {
  long i = (long)blockIdx.x * blockDim.x + threadIdx.x;
  if (i < n) p[i] = v;
}

// convert weight [128 x 128] row-major (k,n) -> transposed bf16 Wt[n][k]
__global__ void ctw_kernel(const float* __restrict__ W, bf16* __restrict__ Wt) {
  int i = blockIdx.x * blockDim.x + threadIdx.x;   // 16384
  int n = i >> 7, k = i & 127;
  Wt[(long)n * 128 + k] = f2bf(W[(long)k * 128 + n]);
}

// build block-diagonal relation weight, transposed: Wt[c][k]
__global__ void crel_kernel(const float* __restrict__ rel, bf16* __restrict__ Wt) {
  int i = blockIdx.x * blockDim.x + threadIdx.x;   // 16384, i = c*128 + k
  int c = i >> 7, k = i & 127;
  int h = c >> 5, j = c & 31;
  float v = ((k >> 5) == h) ? rel[h * (DH * DH) + (k & 31) * DH + j] : 0.0f;
  Wt[i] = f2bf(v);
}

// gelu(exact) + cvt to bf16
__global__ void gelu_cvt_kernel(const float* __restrict__ in, bf16* __restrict__ out, long n) {
  long i = (long)blockIdx.x * blockDim.x + threadIdx.x;
  if (i >= n) return;
  float x = in[i];
  float g = 0.5f * x * (1.0f + erff(x * 0.70710678118654752f));
  out[i] = f2bf(g);
}

// ---------------------------------------------------------------------------
// Multi-weight WMMA GEMM with LDS-staged weights.
//   For w in [0,NW): out_w[M,128] = A[M,128](bf16) @ Wt_w^T + bias_w
//   A slab (16 rows x 128 cols) is held in registers (4 x v16bf) and reused
//   for all NW weights; weights live in LDS and are read via ds_load_b128.
//   MODE==1 (NW==1 only): out = relu(s*(acc+bias) + (1-s)*xres), s=sigmoid(skip)
// blockDim = 256 (8 waves) -> 128 rows per block, 16 rows per wave.
// ---------------------------------------------------------------------------
struct MW {
  const bf16*  W[3];
  const float* bias[3];
  void*        out[3];
  int          obf[3];    // 1 -> store bf16, 0 -> store f32
};

template <int NW, int MODE>
__global__ __launch_bounds__(256) void gemmN_kernel(
    const bf16* __restrict__ A, MW mw, int M,
    const float* __restrict__ xres, const float* __restrict__ skipg, int skip_idx)
{
  __shared__ __align__(16) bf16 Bs[NW * HID * HID];

  // cooperative copy of NW weight matrices (32KB each) into LDS, 16B chunks
  {
    uint4* d = (uint4*)Bs;
#pragma unroll
    for (int w = 0; w < NW; ++w) {
      const uint4* s = (const uint4*)mw.W[w];
      for (int c = threadIdx.x; c < 2048; c += 256)
        d[w * 2048 + c] = s[c];
    }
  }
  __syncthreads();

  int lane = threadIdx.x & 31;
  int wv   = threadIdx.x >> 5;
  int row0 = (blockIdx.x * 8 + wv) * 16;
  if (row0 < M) {                         // wave-uniform: EXEC all-1s inside
    int r15 = lane & 15;
    int kh  = lane >> 4;                  // half-wave select
    long arow = (long)(row0 + r15) * HID;

    // load the full 16x128 A slab into registers (A-layout per ISA table)
    v16bf a[4];
#pragma unroll
    for (int ks = 0; ks < 4; ++ks) {
      const bf16* ap = A + arow + ks * 32 + kh * 8;
      v8bf alo = *(const v8bf*)ap;
      v8bf ahi = *(const v8bf*)(ap + 16);
#pragma unroll
      for (int i = 0; i < 8; ++i) { a[ks][i] = alo[i]; a[ks][8 + i] = ahi[i]; }
    }

    int rbase = row0 + (kh ? 8 : 0);
    float sg = 0.f;
    if (MODE == 1) sg = 1.0f / (1.0f + expf(-skipg[skip_idx]));

#pragma unroll
    for (int w = 0; w < NW; ++w) {
      const bf16* Bw = Bs + w * (HID * HID);
      v8f acc[8];
#pragma unroll
      for (int t = 0; t < 8; ++t) acc[t] = (v8f){0.f,0.f,0.f,0.f,0.f,0.f,0.f,0.f};

#pragma unroll
      for (int ks = 0; ks < 4; ++ks) {
#pragma unroll
        for (int nt = 0; nt < 8; ++nt) {
          // B tile 32x16: lane holds col=lane&15, K = ks*32 + kh*8 + [0..7] and ks*32 + 16 + kh*8 + [0..7]
          const bf16* bp = Bw + (nt * 16 + r15) * HID + ks * 32 + kh * 8;
          v8bf blo = *(const v8bf*)bp;
          v8bf bhi = *(const v8bf*)(bp + 16);
          v16bf b;
#pragma unroll
          for (int i = 0; i < 8; ++i) { b[i] = blo[i]; b[8 + i] = bhi[i]; }
          acc[nt] = __builtin_amdgcn_wmma_f32_16x16x32_bf16(
              false, a[ks], false, b, (short)0, acc[nt], false, false);
        }
      }

      const float* bias = mw.bias[w];
      void* outp = mw.out[w];
      int obf = mw.obf[w];
#pragma unroll
      for (int nt = 0; nt < 8; ++nt) {
        int col = nt * 16 + r15;
        float bv = bias ? bias[col] : 0.0f;
#pragma unroll
        for (int rr = 0; rr < 8; ++rr) {
          long idx = (long)(rbase + rr) * HID + col;
          float v = acc[nt][rr] + bv;
          if (MODE == 1) {
            v = sg * v + (1.0f - sg) * xres[idx];
            v = v > 0.0f ? v : 0.0f;
          }
          if (obf) ((bf16*)outp)[idx] = f2bf(v);
          else     ((float*)outp)[idx] = v;
        }
      }
    }
  }
}

// ---------------------------------------------------------------------------
// Edge phase (per edge type). One thread per (edge, head).
// ---------------------------------------------------------------------------
__global__ void edge_alpha_kernel(const int* __restrict__ ei, const float* __restrict__ qbase,
                                  const float* __restrict__ krel, const float* __restrict__ prel4,
                                  float* __restrict__ alpha, unsigned int* __restrict__ menc, int E)
{
  int i = blockIdx.x * blockDim.x + threadIdx.x;
  if (i >= E * HEADS) return;
  int eidx = i >> 2, h = i & 3;
  int src = ei[eidx], dst = ei[E + eidx];
  const float4* qr = (const float4*)(qbase + (long)dst * HID + h * DH);
  const float4* kr = (const float4*)(krel  + (long)src * HID + h * DH);
  float s = 0.0f;
#pragma unroll
  for (int j = 0; j < 8; ++j) {
    float4 a = qr[j], b = kr[j];
    s += a.x * b.x + a.y * b.y + a.z * b.z + a.w * b.w;
  }
  s *= prel4[h] * 0.17677669529663689f;        // 1/sqrt(32)
  alpha[i] = s;
  unsigned int nv = __float_as_uint(s);
  nv = (nv & 0x80000000u) ? ~nv : (nv | 0x80000000u);   // order-preserving encoding
  atomicMax(&menc[(long)dst * HEADS + h], nv);
}

__global__ void edge_expsum_kernel(float* __restrict__ alpha, const unsigned int* __restrict__ menc,
                                   float* __restrict__ ssum, const int* __restrict__ ei, int E)
{
  int i = blockIdx.x * blockDim.x + threadIdx.x;
  if (i >= E * HEADS) return;
  int eidx = i >> 2, h = i & 3;
  int dst = ei[E + eidx];
  float m = decode_maxenc(menc[(long)dst * HEADS + h]);
  float ex = expf(alpha[i] - m);
  alpha[i] = ex;
  atomicAdd(&ssum[(long)dst * HEADS + h], ex);
}

__global__ void edge_scatter_kernel(const float* __restrict__ alpha, const float* __restrict__ ssum,
                                    const float* __restrict__ vrel, const int* __restrict__ ei,
                                    float* __restrict__ agg, int E)
{
  int i = blockIdx.x * blockDim.x + threadIdx.x;
  if (i >= E * HEADS) return;
  int eidx = i >> 2, h = i & 3;
  int src = ei[eidx], dst = ei[E + eidx];
  float a = alpha[i] / (ssum[(long)dst * HEADS + h] + 1e-16f);
  const float4* vr = (const float4*)(vrel + (long)src * HID + h * DH);
  float* ag = agg + (long)dst * HID + h * DH;
#pragma unroll
  for (int j = 0; j < 8; ++j) {
    float4 v = vr[j];
    atomicAdd(&ag[4 * j + 0], a * v.x);
    atomicAdd(&ag[4 * j + 1], a * v.y);
    atomicAdd(&ag[4 * j + 2], a * v.z);
    atomicAdd(&ag[4 * j + 3], a * v.w);
  }
}

// logits = x0 @ lin_w[128,3] + lin_b
__global__ void lin_head_kernel(const float* __restrict__ x0, const float* __restrict__ w,
                                const float* __restrict__ b, float* __restrict__ out, int n)
{
  int i = blockIdx.x * blockDim.x + threadIdx.x;
  if (i >= n) return;
  const float* xr = x0 + (long)i * HID;
  float a0 = b[0], a1 = b[1], a2 = b[2];
#pragma unroll 8
  for (int k = 0; k < HID; ++k) {
    float xv = xr[k];
    a0 += xv * w[k * 3 + 0];
    a1 += xv * w[k * 3 + 1];
    a2 += xv * w[k * 3 + 2];
  }
  out[(long)i * 3 + 0] = a0;
  out[(long)i * 3 + 1] = a1;
  out[(long)i * 3 + 2] = a2;
}

// ---------------------------------------------------------------------------
// Host orchestration
// ---------------------------------------------------------------------------
extern "C" void kernel_launch(void* const* d_in, const int* in_sizes, int n_in,
                              void* d_out, int out_size, void* d_ws, size_t ws_size,
                              hipStream_t stream) {
  (void)in_sizes; (void)n_in; (void)out_size; (void)ws_size;

  const int  Nn[3]      = {100000, 150000, 50000};
  const long typeOff[3] = {0, 100000, 250000};     // node-major offsets
  const long TN = 300000;                          // total nodes
  const int  Eg = 200000;
  const int  st_[4] = {1, 2, 0, 0};
  const int  dt_[4] = {0, 0, 1, 2};
  const long relOff[4] = {0, 150000, 200000, 300000};  // row offsets per edge type
  const long RN = 400000;

  // ---- workspace layout ----
  size_t off = 0;
  auto alloc = [&](size_t bytes) -> char* {
    off = (off + 255) & ~(size_t)255;
    char* p = (char*)d_ws + off;
    off += bytes;
    return p;
  };
  bf16*  xb     = (bf16*) alloc(TN * HID * sizeof(bf16));
  float* qf     = (float*)alloc(TN * HID * sizeof(float));
  bf16*  kbf    = (bf16*) alloc(TN * HID * sizeof(bf16));
  bf16*  vbf    = (bf16*) alloc(TN * HID * sizeof(bf16));
  float* karel  = (float*)alloc(RN * HID * sizeof(float));
  float* vmrel  = (float*)alloc(RN * HID * sizeof(float));
  float* alphab = (float*)alloc(4L * Eg * HEADS * sizeof(float));
  unsigned int* menc = (unsigned int*)alloc(TN * HEADS * sizeof(unsigned int));
  float* ssum   = (float*)alloc(TN * HEADS * sizeof(float));
  float* aggf   = (float*)alloc(TN * HID * sizeof(float));
  bf16*  gbb    = (bf16*) alloc(TN * HID * sizeof(bf16));
  float* xmid   = (float*)alloc(TN * HID * sizeof(float));
  bf16*  wtpool = (bf16*) alloc(40L * HID * HID * sizeof(bf16));

  auto WT    = [&](int L, int role, int t) { return wtpool + ((long)(L * 4 + role) * 3 + t) * (HID * HID); };
  auto WTREL = [&](int L, int am, int e)   { return wtpool + (24L + (L * 2 + am) * 4 + e) * (HID * HID); };

  auto cdiv = [](long a, long b) { return (int)((a + b - 1) / b); };

  // ---- convert all weights once (transposed bf16 + block-diag relation mats) ----
  for (int L = 0; L < 2; ++L) {
    int base = L ? 19 : 7;
    for (int role = 0; role < 4; ++role) {
      const float* W = (const float*)d_in[base + role];        // [3,128,128]
      for (int t = 0; t < 3; ++t)
        ctw_kernel<<<64, 256, 0, stream>>>(W + (long)t * HID * HID, WT(L, role, t));
    }
    const float* arel = (const float*)d_in[L ? 27 : 15];       // [4,4,32,32]
    const float* mrel = (const float*)d_in[L ? 28 : 16];
    for (int e = 0; e < 4; ++e) {
      crel_kernel<<<64, 256, 0, stream>>>(arel + (long)e * HEADS * DH * DH, WTREL(L, 0, e));
      crel_kernel<<<64, 256, 0, stream>>>(mrel + (long)e * HEADS * DH * DH, WTREL(L, 1, e));
    }
  }

  // ---- one HGT layer ----
  auto run_layer = [&](int L, const float* const xin[3], float* const xout[3]) {
    int base = L ? 19 : 7;
    const float* prel  = (const float*)d_in[L ? 29 : 17];      // [4,4]
    const float* skipg = (const float*)d_in[L ? 30 : 18];      // [3]

    // bf16 copy of inputs
    for (int t = 0; t < 3; ++t) {
      long n = (long)Nn[t] * HID;
      cvt_f32_bf16_kernel<<<cdiv(n, 256), 256, 0, stream>>>(xin[t], xb + typeOff[t] * HID, n);
    }

    // fused K/Q/V projections: A read once, 3 weights from LDS.
    // k,v -> bf16 (feed relation GEMMs), q -> f32 (edge dot products)
    for (int t = 0; t < 3; ++t) {
      MW m{};
      m.W[0] = WT(L, 0, t); m.bias[0] = (const float*)d_in[base + 4] + t * HID;
      m.out[0] = kbf + typeOff[t] * HID; m.obf[0] = 1;
      m.W[1] = WT(L, 1, t); m.bias[1] = (const float*)d_in[base + 5] + t * HID;
      m.out[1] = qf + typeOff[t] * HID;  m.obf[1] = 0;
      m.W[2] = WT(L, 2, t); m.bias[2] = (const float*)d_in[base + 6] + t * HID;
      m.out[2] = vbf + typeOff[t] * HID; m.obf[2] = 1;
      gemmN_kernel<3, 0><<<cdiv(Nn[t], 128), 256, 0, stream>>>(
          xb + typeOff[t] * HID, m, Nn[t], nullptr, nullptr, 0);
    }

    // relation transforms (block-diagonal weights).
    // e0 (src=1) and e1 (src=2) single; e2+e3 share src type 0 -> fused pair.
    for (int am = 0; am < 2; ++am) {
      const bf16* Abase = am ? vbf : kbf;
      float* Obase = am ? vmrel : karel;
      for (int e = 0; e < 2; ++e) {               // e0, e1
        int st = st_[e];
        MW m{};
        m.W[0] = WTREL(L, am, e); m.out[0] = Obase + relOff[e] * HID; m.obf[0] = 0;
        gemmN_kernel<1, 0><<<cdiv(Nn[st], 128), 256, 0, stream>>>(
            Abase + typeOff[st] * HID, m, Nn[st], nullptr, nullptr, 0);
      }
      {                                           // e2 + e3 fused (src type 0)
        MW m{};
        m.W[0] = WTREL(L, am, 2); m.out[0] = Obase + relOff[2] * HID; m.obf[0] = 0;
        m.W[1] = WTREL(L, am, 3); m.out[1] = Obase + relOff[3] * HID; m.obf[1] = 0;
        gemmN_kernel<2, 0><<<cdiv(Nn[0], 128), 256, 0, stream>>>(
            Abase + typeOff[0] * HID, m, Nn[0], nullptr, nullptr, 0);
      }
    }

    // init softmax state (-inf encoded) + sums + agg
    fill_u32_kernel<<<cdiv(TN * HEADS, 256), 256, 0, stream>>>(menc, 0x007FFFFFu, TN * HEADS);
    fill_u32_kernel<<<cdiv(TN * HEADS, 256), 256, 0, stream>>>((unsigned int*)ssum, 0u, TN * HEADS);
    fill_u32_kernel<<<cdiv(TN * HID, 256), 256, 0, stream>>>((unsigned int*)aggf, 0u, TN * HID);

    int egrid = cdiv((long)Eg * HEADS, 256);
    for (int e = 0; e < 4; ++e) {
      const int* ei = (const int*)d_in[3 + e];
      int dt = dt_[e];
      edge_alpha_kernel<<<egrid, 256, 0, stream>>>(ei, qf + typeOff[dt] * HID,
          karel + relOff[e] * HID, prel + e * HEADS,
          alphab + (long)e * Eg * HEADS, menc + typeOff[dt] * HEADS, Eg);
    }
    for (int e = 0; e < 4; ++e) {
      const int* ei = (const int*)d_in[3 + e];
      int dt = dt_[e];
      edge_expsum_kernel<<<egrid, 256, 0, stream>>>(alphab + (long)e * Eg * HEADS,
          menc + typeOff[dt] * HEADS, ssum + typeOff[dt] * HEADS, ei, Eg);
    }
    for (int e = 0; e < 4; ++e) {
      const int* ei = (const int*)d_in[3 + e];
      int dt = dt_[e];
      edge_scatter_kernel<<<egrid, 256, 0, stream>>>(alphab + (long)e * Eg * HEADS,
          ssum + typeOff[dt] * HEADS, vmrel + relOff[e] * HID, ei,
          aggf + typeOff[dt] * HID, Eg);
    }

    // gelu -> bf16, then output projection with fused sigmoid-skip + relu
    for (int t = 0; t < 3; ++t) {
      long n = (long)Nn[t] * HID;
      gelu_cvt_kernel<<<cdiv(n, 256), 256, 0, stream>>>(aggf + typeOff[t] * HID,
                                                        gbb + typeOff[t] * HID, n);
      MW m{};
      m.W[0] = WT(L, 3, t); m.bias[0] = (const float*)d_in[base + 7] + t * HID;
      m.out[0] = xout[t]; m.obf[0] = 0;
      gemmN_kernel<1, 1><<<cdiv(Nn[t], 128), 256, 0, stream>>>(
          gbb + typeOff[t] * HID, m, Nn[t], xin[t], skipg, t);
    }
  };

  float* d_outf = (float*)d_out;
  const long o_x0 = 300000;                                 // after logits [100000,3]
  const long o_x1 = o_x0 + (long)Nn[0] * HID;
  const long o_x2 = o_x1 + (long)Nn[1] * HID;

  const float* xin1[3] = {(const float*)d_in[0], (const float*)d_in[1], (const float*)d_in[2]};
  float* xmid_p[3] = {xmid + typeOff[0] * HID, xmid + typeOff[1] * HID, xmid + typeOff[2] * HID};
  float* xout2[3]  = {d_outf + o_x0, d_outf + o_x1, d_outf + o_x2};

  run_layer(0, xin1, xmid_p);
  const float* xin2[3] = {xmid_p[0], xmid_p[1], xmid_p[2]};
  run_layer(1, xin2, xout2);

  // logits on claim nodes
  lin_head_kernel<<<cdiv(Nn[0], 256), 256, 0, stream>>>(xout2[0],
      (const float*)d_in[31], (const float*)d_in[32], d_outf, Nn[0]);
}
